// MultiHeadSelfAttention_36172214567216
// MI455X (gfx1250) — compile-verified, matrix operand order checked by reading, not run on hardware
//
#include <hip/hip_runtime.h>

// MI455X / gfx1250, wave32. All matmuls via v_wmma_f32_16x16x32_bf16.
//
// Pipeline:
//   1) qkv_gemm_kernel : qkv = x @ w_qkv + b_qkv  -> q/k/v bf16 workspace [b,h,t,d]
//   2) attn_kernel     : flash-style online-softmax attention -> o bf16 [b*t, c]
//   3) proj_gemm_kernel: out = o @ w_proj + b_proj (fp32)
//
// Data movement:
//   - Every WMMA fragment load is 2x ds_load_b128 (B operands stored transposed
//     in LDS at fill time).
//   - Pure-bf16 tile fills use GLOBAL_LOAD_ASYNC_TO_LDS_B128 (ASYNCcnt), double-
//     buffered with partial waits (s_wait_asynccnt 2) so chunk i+1's DMA overlaps
//     chunk i's WMMAs. ASYNCcnt completes in order, so waiting "<= in-flight of
//     next chunk" retires exactly the previous chunk.
//   - f32->bf16 fills are register-staged: next tile's global loads + converts
//     issue in the compute shadow of the current tile.
//
// Workspace (bf16 = ushort): 4 segments of B*T*C = 4096*1024 elems = 33.6 MB.

typedef __attribute__((ext_vector_type(16))) __bf16         v16bf;
typedef __attribute__((ext_vector_type(16))) unsigned short v16u16;
typedef __attribute__((ext_vector_type(8)))  unsigned short v8u16;
typedef __attribute__((ext_vector_type(8)))  float          v8f;

// Exact pointee type of the async-LDS builtin params (probe-verified):
// int __attribute__((vector_size(16))), AS1 global side / AS3 LDS side.
typedef int v4i_vs __attribute__((vector_size(16)));

#define B_ 2
#define T_ 2048
#define C_ 1024
#define H_ 16
#define D_ 64

#if defined(__HIP_DEVICE_COMPILE__) && __has_builtin(__builtin_amdgcn_global_load_async_to_lds_b128)
#define HAVE_ASYNC_LDS 1
#else
#define HAVE_ASYNC_LDS 0
#endif

// f32 -> bf16, round-to-nearest-even (bit trick)
__device__ __forceinline__ unsigned short f2bf(float f) {
  unsigned int u = __float_as_uint(f);
  u += 0x7FFFu + ((u >> 16) & 1u);
  return (unsigned short)(u >> 16);
}

// Load 16 contiguous f32 and pack to 8 dwords of bf16 pairs.
__device__ __forceinline__ void loadcvt16(const float* __restrict__ src, unsigned int* pk) {
#pragma unroll
  for (int i = 0; i < 16; i += 2)
    pk[i >> 1] = (unsigned int)f2bf(src[i]) | ((unsigned int)f2bf(src[i + 1]) << 16);
}

__device__ __forceinline__ v8f wmma_bf16(v16u16 a, v16u16 b, v8f c) {
  return __builtin_amdgcn_wmma_f32_16x16x32_bf16(
      false, __builtin_bit_cast(v16bf, a),
      false, __builtin_bit_cast(v16bf, b),
      (short)0, c, false, false);
}

// 16-byte async global->LDS copy (per-lane addresses), ASYNCcnt-tracked.
__device__ __forceinline__ void cp16_async(const unsigned short* g, unsigned short* l) {
#if HAVE_ASYNC_LDS
  __builtin_amdgcn_global_load_async_to_lds_b128(
      (v4i_vs __attribute__((address_space(1)))*)g,
      (v4i_vs __attribute__((address_space(3)))*)l, 0, 0);
#else
  *(v8u16*)l = *(const v8u16*)g;
#endif
}

// Wait until at most N async global->LDS ops remain in flight (in-order counter).
template <int N>
__device__ __forceinline__ void wait_async() {
#if HAVE_ASYNC_LDS
#if __has_builtin(__builtin_amdgcn_s_wait_asynccnt)
  __builtin_amdgcn_s_wait_asynccnt(N);
#else
  asm volatile("s_wait_asynccnt %0" ::"i"(N) : "memory");
#endif
#endif
}

// A-layout fragment (16x32, 16-bit): lane m = lane&15 picks row, g = lane>>4 picks
// K-halves. elems 0..7 : K = g*8+0..7 ; elems 8..15 : K = 16+g*8+0..7.
// Source is row-major ushort with 'stride' elements per row (stride % 8 == 0).
// Used for BOTH operands: B tiles are stored transposed in LDS so that the
// B-fragment (lane n = column) is also a contiguous row-major load.
__device__ __forceinline__ v16u16 load_frag_rowmajor(const unsigned short* s, int stride,
                                                     int r0, int k0, int lane) {
  int m = lane & 15, g = lane >> 4;
  const unsigned short* p = s + (r0 + m) * stride + k0 + g * 8;
  v8u16 lo = *(const v8u16*)(p);
  v8u16 hi = *(const v8u16*)(p + 16);
  return __builtin_shufflevector(lo, hi, 0, 1, 2, 3, 4, 5, 6, 7,
                                 8, 9, 10, 11, 12, 13, 14, 15);
}

// ---------------------------------------------------------------------------
// Kernel 1: qkv = x[4096,1024] @ w_qkv[1024,3072] + b_qkv, scatter to q/k/v ws.
// Block: 128 thr (4 waves) computes 64x64 tile; wave w owns rows 16w..16w+15.
// Register-staged pipeline: next k-tile loads/converts overlap current WMMAs.
// ---------------------------------------------------------------------------
__global__ __launch_bounds__(128) void qkv_gemm_kernel(
    const float* __restrict__ x, const float* __restrict__ w,
    const float* __restrict__ bias, unsigned short* __restrict__ qkv_ws) {
  __shared__ __attribute__((aligned(16))) unsigned short sA[64 * 40];   // [row][k]
  __shared__ __attribute__((aligned(16))) unsigned short sBt[64 * 40];  // [n][k] (transposed)
  const int tid = threadIdx.x;
  const int lane = tid & 31, wave = tid >> 5;
  const int n0B = blockIdx.x * 64;
  const int r0B = blockIdx.y * 64;

  const int rA = tid >> 1, hfA = tid & 1;   // A-tile fill mapping
  const int krB = tid >> 2, qqB = tid & 3;  // B-tile fill mapping
  const float* srcA = x + (size_t)(r0B + rA) * C_ + hfA * 16;
  const float* srcB = w + (size_t)krB * 3072 + n0B + qqB * 16;

  v8f acc[4] = {{}, {}, {}, {}};
  unsigned int pkA[8], pkB[8];
  loadcvt16(srcA, pkA);
  loadcvt16(srcB, pkB);

  for (int k0 = 0; k0 < C_; k0 += 32) {
    {  // stage -> LDS
      unsigned int* dstA = (unsigned int*)(sA + rA * 40 + hfA * 16);
#pragma unroll
      for (int i = 0; i < 8; ++i) dstA[i] = pkA[i];
#pragma unroll
      for (int i = 0; i < 8; ++i) {
        sBt[(qqB * 16 + 2 * i) * 40 + krB]     = (unsigned short)(pkB[i] & 0xFFFFu);
        sBt[(qqB * 16 + 2 * i + 1) * 40 + krB] = (unsigned short)(pkB[i] >> 16);
      }
    }
    __syncthreads();
    if (k0 + 32 < C_) {  // issue next tile's loads in the compute shadow
      loadcvt16(srcA + (k0 + 32), pkA);
      loadcvt16(srcB + (size_t)(k0 + 32) * 3072, pkB);
      __builtin_prefetch(srcA + (k0 + 64), 0, 0);
      __builtin_prefetch(srcB + (size_t)(k0 + 64) * 3072, 0, 0);
    }
    v16u16 a = load_frag_rowmajor(sA, 40, wave * 16, 0, lane);
#pragma unroll
    for (int j = 0; j < 4; ++j) {
      v16u16 b = load_frag_rowmajor(sBt, 40, j * 16, 0, lane);
      acc[j] = wmma_bf16(a, b, acc[j]);
    }
    __syncthreads();
  }

  // Epilogue: C-layout (lane g=lane>>4, n=lane&15; vgpr v -> row v+8g), add bias,
  // scatter bf16 into q/k/v segments laid out [b*16+h][t][d].
  const int g = lane >> 4, n = lane & 15;
#pragma unroll
  for (int j = 0; j < 4; ++j) {
    int gc = n0B + j * 16 + n;
    int which = gc >> 10;  // 0=q, 1=k, 2=v
    int rem = gc & 1023;
    int hh = rem >> 6, dd = rem & 63;
    float bv = bias[gc];
    unsigned short* seg = qkv_ws + (size_t)which * (4096u * 1024u);
#pragma unroll
    for (int v = 0; v < 8; ++v) {
      int r = r0B + wave * 16 + v + 8 * g;
      int bb = r >> 11, tt = r & 2047;
      seg[(((size_t)(bb * 16 + hh)) * 2048 + tt) * 64 + dd] = f2bf(acc[j][v] + bv);
    }
  }
}

// ---------------------------------------------------------------------------
// Kernel 2: flash-style attention. Block = 128 thr (4 waves) handles 64 query
// rows of one (b,h); wave owns a 16-row strip. KV chunk = 32 keys.
// Double-buffered K (async->LDS) and V (reg-staged transpose); partial
// s_wait_asynccnt retires only the previous chunk.
// ---------------------------------------------------------------------------
__global__ __launch_bounds__(128) void attn_kernel(
    const unsigned short* __restrict__ q_ws, const unsigned short* __restrict__ k_ws,
    const unsigned short* __restrict__ v_ws, unsigned short* __restrict__ o_ws) {
  __shared__ __attribute__((aligned(16))) unsigned short sQ[64 * 72];       // [t][d]
  __shared__ __attribute__((aligned(16))) unsigned short sK[2][32 * 72];    // [s][d]
  __shared__ __attribute__((aligned(16))) unsigned short sVt[2][64 * 40];   // [d][s]
  __shared__ __attribute__((aligned(16))) unsigned short sP[4 * 16 * 40];

  const int tid = threadIdx.x;
  const int lane = tid & 31, wave = tid >> 5;
  const int g = lane >> 4, n = lane & 15;
  const int bh = blockIdx.y;       // b*16 + h
  const int q0 = blockIdx.x * 64;  // query row block within head

  const size_t head_base = (size_t)bh * T_ * D_;
  const int rKV = tid >> 2, qqKV = tid & 3;  // K/V chunk fill mapping
  const unsigned short* kbase = k_ws + head_base + (size_t)rKV * D_ + qqKV * 16;
  const unsigned short* vbase = v_ws + head_base + (size_t)rKV * D_ + qqKV * 16;

  {  // Q block 64x64: pure bf16 copy -> async to LDS
    int r = tid >> 1, hf = tid & 1;
    const unsigned short* src = q_ws + head_base + (size_t)(q0 + r) * D_ + hf * 32;
    unsigned short* dst = sQ + r * 72 + hf * 32;
#pragma unroll
    for (int i = 0; i < 4; ++i) cp16_async(src + i * 8, dst + i * 8);
  }

  v8f acc[4] = {{}, {}, {}, {}};
  float mrow[8], lrow[8];
#pragma unroll
  for (int v = 0; v < 8; ++v) { mrow[v] = -1e30f; lrow[v] = 0.0f; }

  unsigned short* sPw = sP + wave * (16 * 40);

  // Prologue: chunk 0 -> buffer 0 (K async; V via regs, stored immediately)
  cp16_async(kbase, sK[0] + rKV * 72 + qqKV * 16);
  cp16_async(kbase + 8, sK[0] + rKV * 72 + qqKV * 16 + 8);
  {
    v8u16 a0 = *(const v8u16*)(vbase);
    v8u16 a1 = *(const v8u16*)(vbase + 8);
#pragma unroll
    for (int i = 0; i < 8; ++i) sVt[0][(qqKV * 16 + i) * 40 + rKV] = a0[i];
#pragma unroll
    for (int i = 0; i < 8; ++i) sVt[0][(qqKV * 16 + 8 + i) * 40 + rKV] = a1[i];
  }

  for (int it = 0; it < T_ / 32; ++it) {
    const int cur = it & 1, nxt = cur ^ 1;
    const bool more = (it + 1) < (T_ / 32);
    v8u16 v0r = {}, v1r = {};
    if (more) {  // issue next chunk: K async into nxt buffer, V into regs
      const unsigned short* ks = kbase + (size_t)(it + 1) * 32 * D_;
      unsigned short* kd = sK[nxt] + rKV * 72 + qqKV * 16;
      cp16_async(ks, kd);
      cp16_async(ks + 8, kd + 8);
      const unsigned short* vs = vbase + (size_t)(it + 1) * 32 * D_;
      v0r = *(const v8u16*)(vs);
      v1r = *(const v8u16*)(vs + 8);
      __builtin_prefetch(ks + 32 * D_, 0, 0);
      __builtin_prefetch(vs + 32 * D_, 0, 0);
    }
    if (more) wait_async<2>();  // retire previous chunk; keep next 2 in flight
    else      wait_async<0>();
    __syncthreads();  // head barrier: sK[cur]/sVt[cur]/sQ visible to all waves

    // S (16x32) = Q_strip(16x64) @ K_chunk(32x64)^T : B(k=d, n=s) = K[s][d],
    // so the B fragment is an A-style (row-major) load from sK with lane-n as row.
    v8f sc[2] = {{}, {}};
#pragma unroll
    for (int kk = 0; kk < 64; kk += 32) {
      v16u16 a = load_frag_rowmajor(sQ, 72, wave * 16, kk, lane);
#pragma unroll
      for (int j = 0; j < 2; ++j) {
        v16u16 b = load_frag_rowmajor(sK[cur], 72, j * 16, kk, lane);
        sc[j] = wmma_bf16(a, b, sc[j]);
      }
    }

    // Online softmax. Row r = v + 8g lives in 16 lanes (same g); reduce with
    // shfl_xor over the 16-lane half-wave.
    float alpha[8];
#pragma unroll
    for (int v = 0; v < 8; ++v) {
      float x0 = sc[0][v] * 0.125f;  // 1/sqrt(64)
      float x1 = sc[1][v] * 0.125f;
      float mx = fmaxf(x0, x1);
#pragma unroll
      for (int off = 1; off < 16; off <<= 1) mx = fmaxf(mx, __shfl_xor(mx, off, 32));
      float mnew = fmaxf(mrow[v], mx);
      float p0 = __expf(x0 - mnew);
      float p1 = __expf(x1 - mnew);
      float rs = p0 + p1;
#pragma unroll
      for (int off = 1; off < 16; off <<= 1) rs += __shfl_xor(rs, off, 32);
      float a_ = __expf(mrow[v] - mnew);
      lrow[v] = lrow[v] * a_ + rs;
      mrow[v] = mnew;
      alpha[v] = a_;
      // C-layout -> A-layout transpose through per-wave LDS scratch
      sPw[(v + 8 * g) * 40 + n] = f2bf(p0);
      sPw[(v + 8 * g) * 40 + 16 + n] = f2bf(p1);
    }
    __syncthreads();  // sP visible

    // Rescale running O, then O += P(16x32) @ V_chunk(32x64)
#pragma unroll
    for (int j = 0; j < 4; ++j)
#pragma unroll
      for (int v = 0; v < 8; ++v) acc[j][v] = acc[j][v] * alpha[v];

    v16u16 pA = load_frag_rowmajor(sPw, 40, 0, 0, lane);
#pragma unroll
    for (int j = 0; j < 4; ++j) {
      v16u16 b = load_frag_rowmajor(sVt[cur], 40, j * 16, 0, lane);
      acc[j] = wmma_bf16(pA, b, acc[j]);
    }
    __syncthreads();  // tail barrier: everyone done with cur buffers

    if (more) {  // transpose-store staged V regs into the nxt buffer
#pragma unroll
      for (int i = 0; i < 8; ++i) sVt[nxt][(qqKV * 16 + i) * 40 + rKV] = v0r[i];
#pragma unroll
      for (int i = 0; i < 8; ++i) sVt[nxt][(qqKV * 16 + 8 + i) * 40 + rKV] = v1r[i];
    }
  }

  // Finalize: O /= l, write bf16 to o_ws [b*T + t][h*64 + d]
  const int b = bh >> 4, h = bh & 15;
#pragma unroll
  for (int v = 0; v < 8; ++v) {
    float inv = 1.0f / lrow[v];
    int t = q0 + wave * 16 + v + 8 * g;
    size_t row = (size_t)(b * T_ + t);
#pragma unroll
    for (int j = 0; j < 4; ++j)
      o_ws[row * C_ + h * 64 + j * 16 + n] = f2bf(acc[j][v] * inv);
  }
}

// ---------------------------------------------------------------------------
// Kernel 3: out[4096,1024] (f32) = o_ws(bf16) @ w_proj[1024,1024] + b_proj
// A tile: async->LDS, double-buffered. B tile: register-staged f32->bf16.
// ---------------------------------------------------------------------------
__global__ __launch_bounds__(128) void proj_gemm_kernel(
    const unsigned short* __restrict__ o_ws, const float* __restrict__ w,
    const float* __restrict__ bias, float* __restrict__ out) {
  __shared__ __attribute__((aligned(16))) unsigned short sA[2][64 * 40];  // [row][k]
  __shared__ __attribute__((aligned(16))) unsigned short sBt[64 * 40];    // [n][k]
  const int tid = threadIdx.x;
  const int lane = tid & 31, wave = tid >> 5;
  const int n0B = blockIdx.x * 64;
  const int r0B = blockIdx.y * 64;

  const int rA = tid >> 1, hfA = tid & 1;
  const int krB = tid >> 2, qqB = tid & 3;
  const unsigned short* srcA = o_ws + (size_t)(r0B + rA) * C_ + hfA * 16;
  const float* srcB = w + (size_t)krB * C_ + n0B + qqB * 16;

  v8f acc[4] = {{}, {}, {}, {}};
  unsigned int pkB[8];
  loadcvt16(srcB, pkB);
  // Prologue: A tile k0=0 -> sA[0]
  cp16_async(srcA, sA[0] + rA * 40 + hfA * 16);
  cp16_async(srcA + 8, sA[0] + rA * 40 + hfA * 16 + 8);

  for (int it = 0; it < C_ / 32; ++it) {
    const int k0 = it * 32;
    const int cur = it & 1, nxt = cur ^ 1;
    const bool more = (k0 + 32) < C_;
    {  // stage B -> LDS (transposed)
#pragma unroll
      for (int i = 0; i < 8; ++i) {
        sBt[(qqB * 16 + 2 * i) * 40 + krB]     = (unsigned short)(pkB[i] & 0xFFFFu);
        sBt[(qqB * 16 + 2 * i + 1) * 40 + krB] = (unsigned short)(pkB[i] >> 16);
      }
    }
    if (more) {  // issue next A tile async into nxt buffer
      const unsigned short* sa = srcA + (k0 + 32);
      unsigned short* da = sA[nxt] + rA * 40 + hfA * 16;
      cp16_async(sa, da);
      cp16_async(sa + 8, da + 8);
      wait_async<2>();
    } else {
      wait_async<0>();
    }
    __syncthreads();
    if (more) {  // next B tile loads/converts in the compute shadow
      loadcvt16(srcB + (size_t)(k0 + 32) * C_, pkB);
      __builtin_prefetch(srcB + (size_t)(k0 + 64) * C_, 0, 0);
    }
    v16u16 a = load_frag_rowmajor(sA[cur], 40, wave * 16, 0, lane);
#pragma unroll
    for (int j = 0; j < 4; ++j) {
      v16u16 b = load_frag_rowmajor(sBt, 40, j * 16, 0, lane);
      acc[j] = wmma_bf16(a, b, acc[j]);
    }
    __syncthreads();
  }

  const int g = lane >> 4, n = lane & 15;
#pragma unroll
  for (int j = 0; j < 4; ++j) {
    int gc = n0B + j * 16 + n;
    float bv = bias[gc];
#pragma unroll
    for (int v = 0; v < 8; ++v) {
      int r = r0B + wave * 16 + v + 8 * g;
      out[(size_t)r * C_ + gc] = acc[j][v] + bv;
    }
  }
}

extern "C" void kernel_launch(void* const* d_in, const int* in_sizes, int n_in,
                              void* d_out, int out_size, void* d_ws, size_t ws_size,
                              hipStream_t stream) {
  (void)in_sizes; (void)n_in; (void)out_size; (void)ws_size;
  const float* x      = (const float*)d_in[0];
  const float* w_qkv  = (const float*)d_in[1];
  const float* b_qkv  = (const float*)d_in[2];
  const float* w_proj = (const float*)d_in[3];
  const float* b_proj = (const float*)d_in[4];
  float* out = (float*)d_out;

  unsigned short* ws = (unsigned short*)d_ws;
  const size_t SEG = (size_t)4096 * 1024;  // B*T*C bf16 elements
  unsigned short* q_ws = ws;               // q,k,v contiguous; kernel1 indexes by segment
  unsigned short* k_ws = ws + SEG;
  unsigned short* v_ws = ws + 2 * SEG;
  unsigned short* o_ws = ws + 3 * SEG;

  qkv_gemm_kernel<<<dim3(48, 64), 128, 0, stream>>>(x, w_qkv, b_qkv, q_ws);
  attn_kernel<<<dim3(32, 32), 128, 0, stream>>>(q_ws, k_ws, v_ws, o_ws);
  proj_gemm_kernel<<<dim3(16, 64), 128, 0, stream>>>(o_ws, w_proj, b_proj, out);
}
